// CrossAttention_20684562498160
// MI455X (gfx1250) — matrix operand order repaired and checked by reading, not compiled, not run on hardware
//
#include <hip/hip_runtime.h>
#include <hip/hip_bf16.h>
#include <stdint.h>

// ---------------------------------------------------------------------------
// CrossAttention for MI455X (gfx1250), bf16 WMMA path.
// B=8, K=1024, H=W=24 -> C=576, NH=6, HD=96.
// ---------------------------------------------------------------------------

typedef __attribute__((ext_vector_type(16))) __bf16 v16bf;
typedef __attribute__((ext_vector_type(8)))  float  v8f;

struct alignas(16) B8 { __bf16 h[8]; };   // 8 bf16 = 16 bytes (global/ds b128)

static constexpr int BB  = 8;
static constexpr int KK  = 1024;   // tokens
static constexpr int CC  = 576;    // channels
static constexpr int NHh = 6;
static constexpr int HDd = 96;

// ---- helper: build A/B fragment (v16bf) from two aligned 8-elem runs -------
__device__ __forceinline__ v16bf frag16(const __bf16* p0, const __bf16* p1) {
  B8 r0 = *(const B8*)p0;
  B8 r1 = *(const B8*)p1;
  v16bf a;
#pragma unroll
  for (int e = 0; e < 8; ++e) { a[e] = r0.h[e]; a[e + 8] = r1.h[e]; }
  return a;
}

// ---------------------------------------------------------------------------
// 0) f32 -> bf16 elementwise convert
// ---------------------------------------------------------------------------
__global__ __launch_bounds__(256) void cvt_f32_bf16(const float* __restrict__ s,
                                                    __bf16* __restrict__ d, int n) {
  int i = blockIdx.x * 256 + threadIdx.x;
  if (i < n) d[i] = (__bf16)s[i];
}

// ---------------------------------------------------------------------------
// 1) Projection GEMM: Y[m,n] = sum_c X[m,c] * W[n,c] + bias[n]   (bf16 in/out)
//    M=8192 (tokens), N=C=576, Kdim=C=576. Workgroup = 64x64 tile, 8 waves.
//    Wave w: m-subtile (w>>1), n-subtiles (w&1)*2 + {0,1}.
//    A frag straight from X (row-major, contiguous c); B frag straight from W
//    (W[n][c] row-major == exactly the B[kd=c, n] layout runs). No LDS needed.
// ---------------------------------------------------------------------------
__global__ __launch_bounds__(256) void proj_gemm(const __bf16* __restrict__ X,
                                                 const __bf16* __restrict__ W,
                                                 const float*  __restrict__ bias,
                                                 __bf16* __restrict__ Y) {
  const int mbase = blockIdx.x * 64;
  const int nbase = blockIdx.y * 64;
  const int tid = threadIdx.x;
  const int w  = tid >> 5;
  const int l  = tid & 31;
  const int hi = l >> 4;
  const int li = l & 15;
  const int mt  = w >> 1;          // 0..3
  const int nt0 = (w & 1) * 2;     // 0 or 2

  const int row = mbase + mt * 16 + li;      // A-matrix row (token), lane-resident
  v8f acc0 = {};
  v8f acc1 = {};

#pragma unroll 2
  for (int cb = 0; cb < CC; cb += 32) {
    // A[m=token, k=c] : elems 0..15 at c = cb + hi*16 + e (contiguous)
    const __bf16* ap = X + (size_t)row * CC + cb + hi * 16;
    v16bf a = frag16(ap, ap + 8);

    // B[k=c, n] = W[n][c] : contiguous 16 c-values at row n, offset cb + hi*16
    const int n0 = nbase + nt0 * 16 + li;
    const __bf16* bp0 = W + (size_t)n0 * CC + cb + hi * 16;
    v16bf b0 = frag16(bp0, bp0 + 8);
    acc0 = __builtin_amdgcn_wmma_f32_16x16x32_bf16(false, a, false, b0,
                                                   (short)0, acc0, false, false);

    const int n1 = n0 + 16;
    const __bf16* bp1 = W + (size_t)n1 * CC + cb + hi * 16;
    v16bf b1 = frag16(bp1, bp1 + 8);
    acc1 = __builtin_amdgcn_wmma_f32_16x16x32_bf16(false, a, false, b1,
                                                   (short)0, acc1, false, false);
  }

  // bias + store (D layout: row m = vgpr + hi*8, col n = li)
  const int n0 = nbase + nt0 * 16 + li;
  const int n1 = n0 + 16;
  const float bv0 = bias[n0];
  const float bv1 = bias[n1];
#pragma unroll
  for (int j = 0; j < 8; ++j) {
    const int m = mbase + mt * 16 + hi * 8 + j;
    Y[(size_t)m * CC + n0] = (__bf16)(acc0[j] + bv0);
    Y[(size_t)m * CC + n1] = (__bf16)(acc1[j] + bv1);
  }
}

// ---------------------------------------------------------------------------
// 2) Attention per (b, head, 32-row q-tile).
//    View q/k/v flat buffers as [b][h][d=96][k=1024] (the reshape scramble).
//    LDS: sc[32][1024] f32 (128K) + Pb[32][1024] bf16 (64K)
//       + Qt[32][96] bf16 (6K) + Kt[128][96] bf16 (24K) = 222 KB of 320 KB.
//    Phase A: scores = Q^T K  (3 bf16 WMMAs per 16x16 tile, d = 3x32),
//             with global_prefetch of the next K chunk overlapping compute.
//    Phase B: rowwise softmax; normalize pass emits bf16 P directly.
//    Phase C: ctx^T[q][d] = P · V^T ; pure ds_b128/global_b128 + WMMA loop.
// ---------------------------------------------------------------------------
__global__ __launch_bounds__(256) void attn_kernel(const __bf16* __restrict__ Q,
                                                   const __bf16* __restrict__ Km,
                                                   const __bf16* __restrict__ V,
                                                   float* __restrict__ out) {
  extern __shared__ char smem[];
  float*  sc = (float*)smem;                                  // 131072 B
  __bf16* Pb = (__bf16*)(smem + 131072);                      //  65536 B
  __bf16* Qt = (__bf16*)(smem + 131072 + 65536);              //   6144 B
  __bf16* Kt = (__bf16*)(smem + 131072 + 65536 + 6144);       //  24576 B

  const int b  = blockIdx.z;
  const int h  = blockIdx.y;
  const int qb = blockIdx.x * 32;

  const size_t headOff = (size_t)b * KK * CC + (size_t)h * HDd * KK;
  const __bf16* Qh = Q  + headOff;   // [d][k], k contiguous
  const __bf16* Kh = Km + headOff;
  const __bf16* Vh = V  + headOff;

  const int tid = threadIdx.x;
  const int w  = tid >> 5;
  const int l  = tid & 31;
  const int hi = l >> 4;
  const int li = l & 15;
  const int qt = w & 1;               // q-subtile 0/1

  // ---- stage Q tile transposed into LDS: Qt[q][d] ----
  for (int i = tid; i < 32 * 96; i += 256) {
    const int q = i & 31;             // contiguous in global
    const int d = i >> 5;
    Qt[q * 96 + d] = Qh[(size_t)d * KK + qb + q];
  }
  __syncthreads();

  // ---- preload the 3 Q^T A-fragments for this wave's q-subtile ----
  v16bf aq[3];
  {
    const int q = qt * 16 + li;
#pragma unroll
    for (int dc = 0; dc < 3; ++dc) {
      const __bf16* ap = Qt + q * 96 + dc * 32 + hi * 16;
      aq[dc] = frag16(ap, ap + 8);
    }
  }

  // ---- Phase A: scores, 128 keys per chunk ----
  for (int kc = 0; kc < 8; ++kc) {
    __syncthreads();                                   // Kt reuse fence
    for (int i = tid; i < 128 * 96; i += 256) {
      const int kl = i & 127;                          // contiguous in global
      const int d  = i >> 7;
      Kt[kl * 96 + d] = Kh[(size_t)d * KK + kc * 128 + kl];
    }
    // Warm WGP$/L2 for the NEXT chunk while this one computes
    // (96 rows x 256B = 4 cachelines/row). Emits global_prefetch_b8.
    if (kc + 1 < 8) {
      for (int i = tid; i < 96 * 4; i += 256) {
        const int d   = i >> 2;
        const int off = (i & 3) * 64;
        __builtin_prefetch((const char*)(Kh + (size_t)d * KK + (kc + 1) * 128) + off,
                           0, 3);
      }
    }
    __syncthreads();

#pragma unroll
    for (int t = 0; t < 2; ++t) {
      const int nt = (w >> 1) * 2 + t;                 // 0..7 key-subtile
      v8f acc = {};
#pragma unroll
      for (int dc = 0; dc < 3; ++dc) {
        const int kl = nt * 16 + li;
        const __bf16* bp = Kt + kl * 96 + dc * 32 + hi * 16;
        v16bf bf = frag16(bp, bp + 8);
        acc = __builtin_amdgcn_wmma_f32_16x16x32_bf16(false, aq[dc], false, bf,
                                                      (short)0, acc, false, false);
      }
      const int col = kc * 128 + nt * 16 + li;
#pragma unroll
      for (int j = 0; j < 8; ++j)
        sc[(qt * 16 + hi * 8 + j) * 1024 + col] = acc[j];
    }
  }
  __syncthreads();

  // ---- Phase B: softmax over each of the 32 rows (no 1/sqrt(d) scale!).
  //      Normalize pass converts straight to bf16 P (done once per element,
  //      so the PV loop never touches f32 scores again). ----
  {
    const int r   = tid >> 3;         // row 0..31
    const int seg = tid & 7;          // 8 lanes per row, 128 elems each
    float*  row  = sc + r * 1024 + seg * 128;
    __bf16* prow = Pb + r * 1024 + seg * 128;
    float mx = -1e30f;
#pragma unroll 4
    for (int i = 0; i < 128; ++i) mx = fmaxf(mx, row[i]);
#pragma unroll
    for (int o = 1; o < 8; o <<= 1) mx = fmaxf(mx, __shfl_xor(mx, o, 8));
    float s = 0.f;
#pragma unroll 4
    for (int i = 0; i < 128; ++i) { float e = __expf(row[i] - mx); row[i] = e; s += e; }
#pragma unroll
    for (int o = 1; o < 8; o <<= 1) s += __shfl_xor(s, o, 8);
    const float inv = 1.0f / s;
#pragma unroll 4
    for (int i = 0; i < 128; ++i) prow[i] = (__bf16)(row[i] * inv);
  }
  __syncthreads();

  // ---- Phase C: ctx^T[q][d] = P * V^T ----
  // 12 output tiles (2 q x 6 d). Wave w owns (qt, dt0=w>>1); waves 0..3 also
  // own (qt, dt1=dt0+4). Wave-uniform branch keeps EXEC full for WMMA.
  const int  dt0 = w >> 1;
  const int  dt1 = dt0 + 4;
  const bool two = (w < 4);
  v8f c0 = {};
  v8f c1 = {};

  for (int kb = 0; kb < 32; ++kb) {
    // A = P[q, k] : two aligned ds b128 loads, already bf16
    const int q = qt * 16 + li;
    const __bf16* pp = Pb + q * 1024 + kb * 32 + hi * 16;
    v16bf a = frag16(pp, pp + 8);

    // B[k, n=d] = V[d][k] : contiguous k-run straight from global
    const int d0 = dt0 * 16 + li;
    const __bf16* vp0 = Vh + (size_t)d0 * KK + kb * 32 + hi * 16;
    v16bf b0 = frag16(vp0, vp0 + 8);
    c0 = __builtin_amdgcn_wmma_f32_16x16x32_bf16(false, a, false, b0,
                                                 (short)0, c0, false, false);
    if (two) {
      const int d1 = dt1 * 16 + li;
      const __bf16* vp1 = Vh + (size_t)d1 * KK + kb * 32 + hi * 16;
      v16bf b1 = frag16(vp1, vp1 + 8);
      c1 = __builtin_amdgcn_wmma_f32_16x16x32_bf16(false, a, false, b1,
                                                   (short)0, c1, false, false);
    }
  }

  // ---- store: out flat index per batch = h*HD*K + d*K + q ----
  float* outB = out + (size_t)b * KK * CC + (size_t)h * HDd * KK;
#pragma unroll
  for (int j = 0; j < 8; ++j) {
    const int qg = qb + qt * 16 + hi * 8 + j;
    const int d0 = dt0 * 16 + li;
    outB[(size_t)d0 * KK + qg] = c0[j];
  }
  if (two) {
#pragma unroll
    for (int j = 0; j < 8; ++j) {
      const int qg = qb + qt * 16 + hi * 8 + j;
      const int d1 = dt1 * 16 + li;
      outB[(size_t)d1 * KK + qg] = c1[j];
    }
  }
}

// ---------------------------------------------------------------------------
// Host launcher
// ---------------------------------------------------------------------------
extern "C" void kernel_launch(void* const* d_in, const int* in_sizes, int n_in,
                              void* d_out, int out_size, void* d_ws, size_t ws_size,
                              hipStream_t stream) {
  (void)in_sizes; (void)n_in; (void)out_size; (void)ws_size;

  const float* x1 = (const float*)d_in[0];
  const float* x2 = (const float*)d_in[1];
  const float* Wq = (const float*)d_in[2];
  const float* bq = (const float*)d_in[3];
  const float* Wk = (const float*)d_in[4];
  const float* bk = (const float*)d_in[5];
  const float* Wv = (const float*)d_in[6];
  const float* bv = (const float*)d_in[7];
  float* out = (float*)d_out;

  const size_t NX = (size_t)BB * KK * CC;   // 4,718,592 elements
  const size_t NW = (size_t)CC * CC;        //   331,776 elements

  char* ws = (char*)d_ws;
  __bf16* X1b = (__bf16*)ws;  ws += NX * 2;
  __bf16* X2b = (__bf16*)ws;  ws += NX * 2;
  __bf16* Wqb = (__bf16*)ws;  ws += NW * 2;
  __bf16* Wkb = (__bf16*)ws;  ws += NW * 2;
  __bf16* Wvb = (__bf16*)ws;  ws += NW * 2;
  __bf16* Qb  = (__bf16*)ws;  ws += NX * 2;
  __bf16* Kb  = (__bf16*)ws;  ws += NX * 2;
  __bf16* Vb  = (__bf16*)ws;  ws += NX * 2;

  // 0) converts
  cvt_f32_bf16<<<(int)((NX + 255) / 256), 256, 0, stream>>>(x1, X1b, (int)NX);
  cvt_f32_bf16<<<(int)((NX + 255) / 256), 256, 0, stream>>>(x2, X2b, (int)NX);
  cvt_f32_bf16<<<(int)((NW + 255) / 256), 256, 0, stream>>>(Wq, Wqb, (int)NW);
  cvt_f32_bf16<<<(int)((NW + 255) / 256), 256, 0, stream>>>(Wk, Wkb, (int)NW);
  cvt_f32_bf16<<<(int)((NW + 255) / 256), 256, 0, stream>>>(Wv, Wvb, (int)NW);

  // 1) projections: M=8192 tokens x N=576, 64x64 tiles
  dim3 pg(8192 / 64, CC / 64, 1);
  proj_gemm<<<pg, 256, 0, stream>>>(X1b, Wqb, bq, Qb);
  proj_gemm<<<pg, 256, 0, stream>>>(X2b, Wkb, bk, Kb);
  proj_gemm<<<pg, 256, 0, stream>>>(X2b, Wvb, bv, Vb);

  // 2) attention: 222 KB dynamic LDS per workgroup (WGP has 320 KB)
  const int smemBytes = 131072 + 65536 + 6144 + 24576;
  hipFuncSetAttribute(reinterpret_cast<const void*>(attn_kernel),
                      hipFuncAttributeMaxDynamicSharedMemorySize, smemBytes);
  attn_kernel<<<dim3(KK / 32, NHh, BB), 256, smemBytes, stream>>>(Qb, Kb, Vb, out);
}
